// PairwisePotentialsEncoder_34248069218451
// MI455X (gfx1250) — matrix operand order repaired and checked by reading, not compiled, not run on hardware
//
#include <hip/hip_runtime.h>
#include <hip/hip_bf16.h>

typedef _Float16 half8_t __attribute__((ext_vector_type(8)));
typedef _Float16 v16h    __attribute__((ext_vector_type(16)));
typedef float    v8f     __attribute__((ext_vector_type(8)));

#define Bb 2
#define Kc 8
#define Cc 64
#define Hh 96
#define Ww 96
#define HW (Hh * Ww)     /* 9216 */
#define CK (Cc * Kc)     /* 512  */
#define TPW 3            /* tiles per wave in feat_kernel */

// ---------- helpers ----------
static __device__ __forceinline__ v16h pack16(half8_t lo, half8_t hi) {
  v16h r;
#pragma unroll
  for (int i = 0; i < 8; ++i) { r[i] = lo[i]; r[i + 8] = hi[i]; }
  return r;
}

static __device__ __forceinline__ v16h ld_bfrag(const _Float16* p) {
  // 16 contiguous f16 (two b128 loads) -> one B/A run pair
  return pack16(*(const half8_t*)p, *(const half8_t*)(p + 8));
}

static __device__ __forceinline__ v8f wmma16x16x32(v16h a, v16h b, v8f c) {
  // D = A(16x32 f16) * B(32x16 f16) + C(16x16 f32)
  return __builtin_amdgcn_wmma_f32_16x16x32_f16(
      /*neg_a=*/false, a, /*neg_b=*/false, b,
      /*c_mod=*/(short)0, c, /*reuse_a=*/false, /*reuse_b=*/false);
}

// =====================================================================
// Kernel 1: fused  h1 = relu(W1*(x+pe)+b1) ; h2 = W2*h1+b2
// Weights are converted to f16 ONCE per block into LDS (8 KB + 64 KB),
// then every B-fragment is a ds_load_b128 pair. Each of the 4 waves
// processes TPW=3 sixteen-pixel tiles, amortizing the staging 12x.
// Stores h2 (f16) and u = pw[c]*h2 (f16) to workspace.
// =====================================================================
__global__ __launch_bounds__(128) void feat_kernel(
    const float* __restrict__ x,  const float* __restrict__ pe,
    const float* __restrict__ w1, const float* __restrict__ b1,
    const float* __restrict__ w2, const float* __restrict__ b2,
    const float* __restrict__ pw,
    _Float16* __restrict__ h2ws,  _Float16* __restrict__ uws)
{
  extern __shared__ _Float16 sm[];
  _Float16* w1s = sm;                   // 4096  halves ( 8 KB)  [o][c]
  _Float16* w2s = sm + 4096;            // 32768 halves (64 KB)  [o][c]
  _Float16* trs = sm + 4096 + 32768;    // 4096  halves ( 8 KB)  transpose buf

  // ---- cooperative one-time weight conversion fp32 -> f16 into LDS ----
  for (int i = threadIdx.x; i < 64 * 64; i += 128)  w1s[i] = (_Float16)w1[i];
  for (int i = threadIdx.x; i < 512 * 64; i += 128) w2s[i] = (_Float16)w2[i];
  __syncthreads();

  const int lane = threadIdx.x & 31;
  const int wave = threadIdx.x >> 5;
  const int mrow = lane & 15;               // M (rows) for A / N (cols) for B
  const int hi   = (lane >= 16) ? 1 : 0;
  const int bkb  = hi ? 16 : 0;             // B-fragment K base (step 0)
  _Float16* myLds = trs + wave * 16 * Cc;

  for (int it = 0; it < TPW; ++it) {
    const int tile = (blockIdx.x * 4 + wave) * TPW + it;  // 0..1151
    const int b    = (tile * 16) / HW;
    const int pix0 = (tile * 16) % HW;

    // ---- A fragments from x + pe (fp32 gather, cvt f16), K = channel ----
    const int pixl = pix0 + mrow;
    const float* xb  = x  + (size_t)b * Cc * HW + pixl;
    const float* peb = pe + pixl;
    if (it + 1 < TPW) __builtin_prefetch(xb + 16, 0, 0);  // next tile's x
    v16h a0, a1;
    {
      half8_t c0, c1, c2, c3;
#pragma unroll
      for (int i = 0; i < 8; ++i) {
        int k0 = (hi ? 16 : 0)  + i;
        int k1 = (hi ? 24 : 8) + i;
        int k2 = 32 + (hi ? 16 : 0)  + i;
        int k3 = 32 + (hi ? 24 : 8) + i;
        c0[i] = (_Float16)(xb[(size_t)k0 * HW] + peb[(size_t)k0 * HW]);
        c1[i] = (_Float16)(xb[(size_t)k1 * HW] + peb[(size_t)k1 * HW]);
        c2[i] = (_Float16)(xb[(size_t)k2 * HW] + peb[(size_t)k2 * HW]);
        c3[i] = (_Float16)(xb[(size_t)k3 * HW] + peb[(size_t)k3 * HW]);
      }
      a0 = pack16(c0, c1);
      a1 = pack16(c2, c3);
    }

    // ---- GEMM1: h1 = relu(A * W1^T + b1), store f16 to LDS [pixel][o] ----
#pragma unroll
    for (int ot = 0; ot < 4; ++ot) {
      const int o = ot * 16 + mrow;
      const _Float16* wr = w1s + o * Cc;
      v16h bf0 = ld_bfrag(wr + bkb);
      v16h bf1 = ld_bfrag(wr + 32 + bkb);
      v8f acc = {};
      acc = wmma16x16x32(a0, bf0, acc);
      acc = wmma16x16x32(a1, bf1, acc);
      const float bias = b1[o];
#pragma unroll
      for (int r = 0; r < 8; ++r) {
        float v = acc[r] + bias;
        v = v > 0.f ? v : 0.f;
        myLds[(r + hi * 8) * Cc + o] = (_Float16)v;
      }
    }
    asm volatile("s_wait_dscnt 0x0" ::: "memory");

    // ---- reload h1 as A fragments (wave-local ds_load_b128) ----
    {
      const _Float16* lp = myLds + mrow * Cc;
      half8_t c0 = *(const half8_t*)(lp + (hi ? 16 : 0));
      half8_t c1 = *(const half8_t*)(lp + (hi ? 24 : 8));
      half8_t c2 = *(const half8_t*)(lp + 32 + (hi ? 16 : 0));
      half8_t c3 = *(const half8_t*)(lp + 32 + (hi ? 24 : 8));
      a0 = pack16(c0, c1);
      a1 = pack16(c2, c3);
    }

    // ---- GEMM2: h2[pix][o], o in 0..511; emit h2 (f16) and u = pw*h2 ----
    for (int ot = 0; ot < 32; ++ot) {
      const int o = ot * 16 + mrow;
      const _Float16* wr = w2s + o * Cc;
      v16h bf0 = ld_bfrag(wr + bkb);
      v16h bf1 = ld_bfrag(wr + 32 + bkb);
      v8f acc = {};
      acc = wmma16x16x32(a0, bf0, acc);
      acc = wmma16x16x32(a1, bf1, acc);
      const float bias = b2[o];
      const float pwc  = pw[o & 63];
#pragma unroll
      for (int r = 0; r < 8; ++r) {
        const float v   = acc[r] + bias;
        const int   px  = r + hi * 8;
        const size_t ad = ((size_t)b * HW + pix0 + px) * CK + o;
        h2ws[ad] = (_Float16)v;
        uws[ad]  = (_Float16)(pwc * v);
      }
    }
  }
}

// =====================================================================
// Kernel 2: S[b,pix,j] = sum_c pw[c] * h2[j*64+c]^2
// =====================================================================
__global__ __launch_bounds__(256) void s_kernel(
    const _Float16* __restrict__ h2ws, const float* __restrict__ pw,
    float* __restrict__ Sws)
{
  const int idx = blockIdx.x * 256 + threadIdx.x;   // (b*HW+pix)*8 + j
  if (idx >= Bb * HW * Kc) return;
  const _Float16* hp = h2ws + (size_t)idx * Cc;
  float s = 0.f;
#pragma unroll 8
  for (int c = 0; c < Cc; ++c) {
    const float v = (float)hp[c];
    s += pw[c] * v * v;
  }
  Sws[idx] = s;
}

// =====================================================================
// Kernel 3: pairwise potentials.
//   pot[j,k] = S[j] + S'[k] - 2*Gram(u_j, h'_k) + pb ;  out = exp(pot)
// One block = 16 pixels of one row (b,y,x0..x0+15), all 25 offsets.
// A wave's WMMA: rows = (2 px)x(8 j) of u, cols = (2 px)x(8 k) of shifted h.
// Results staged in 96 KB LDS, then fully coalesced writeback.
// =====================================================================
__global__ __launch_bounds__(256) void pot_kernel(
    const _Float16* __restrict__ h2ws, const _Float16* __restrict__ uws,
    const float* __restrict__ Sws, const float* __restrict__ pb,
    float* __restrict__ out)
{
  extern __shared__ float spot[];            // [jk(64)][t24(24)][xl(16)] = 96 KB
  const int blk = blockIdx.x;                // b*576 + y*6 + xt
  const int b   = blk / 576;
  const int rem = blk % 576;
  const int y   = rem / 6;
  const int x0  = (rem % 6) * 16;
  const int lane = threadIdx.x & 31;
  const int wave = threadIdx.x >> 5;
  const int mrow = lane & 15;
  const int hi   = (lane >= 16) ? 1 : 0;
  const size_t baseBP = (size_t)b * HW;
  const float pbv = pb[0];

  // 192 tasks = 24 offsets x 8 pixel-pairs, round-robined over 8 waves
  for (int task = wave; task < 192; task += 8) {
    const int pr  = task & 7;                 // pixel pair -> x = x0+2pr, +1
    const int t24 = task >> 3;                // 0..23 (offset, center skipped)
    const int t   = t24 + (t24 >= 12 ? 1 : 0);
    const int dx  = t / 5 - 2;
    const int dy  = t % 5 - 2;

    // ---- A: u for the two unshifted pixels. row M = (member m)*8 + j ----
    const int am = mrow >> 3;
    const int aj = mrow & 7;
    const _Float16* ap =
        uws + (baseBP + (size_t)y * Ww + (x0 + pr * 2 + am)) * CK + aj * Cc;
    const v16h A0 = pack16(*(const half8_t*)(ap + (hi ? 16 : 0)),
                           *(const half8_t*)(ap + (hi ? 24 : 8)));
    const v16h A1 = pack16(*(const half8_t*)(ap + 32 + (hi ? 16 : 0)),
                           *(const half8_t*)(ap + 32 + (hi ? 24 : 8)));

    // ---- B: shifted h2. col N = (member mn)*8 + k ; roll() indexing ----
    const int mn = mrow >> 3;
    const int k  = mrow & 7;
    int ys = y - dx;                 if (ys < 0) ys += Hh; else if (ys >= Hh) ys -= Hh;
    int xs = x0 + pr * 2 + mn - dy;  if (xs < 0) xs += Ww; else if (xs >= Ww) xs -= Ww;
    const _Float16* bp =
        h2ws + (baseBP + (size_t)ys * Ww + xs) * CK + k * Cc;
    const int kb = hi ? 16 : 0;
    const v16h B0 = ld_bfrag(bp + kb);
    const v16h B1 = ld_bfrag(bp + 32 + kb);

    v8f acc = {};
    acc = wmma16x16x32(A0, B0, acc);
    acc = wmma16x16x32(A1, B1, acc);

    // ---- diagonal 8x8 blocks: lanes 0..7 (member 0), 24..31 (member 1) ----
    const bool useful = (lane < 8) || (lane >= 24);
    if (useful) {
      const int mu = (lane >= 24) ? 1 : 0;    // equals mn on these lanes
      const int ku = lane & 7;                // equals k  on these lanes
      const int xl = pr * 2 + mu;
      const float sshift = Sws[(baseBP + (size_t)ys * Ww + xs) * Kc + ku];
      const float* ssp   = Sws + (baseBP + (size_t)y * Ww + (x0 + xl)) * Kc;
#pragma unroll
      for (int r = 0; r < 8; ++r) {           // r == j for both members
        const float pot = ssp[r] + sshift - 2.f * acc[r] + pbv;
        spot[((r * 8 + ku) * 24 + t24) * 16 + xl] = __expf(pot);
      }
    }
  }

  __syncthreads();

  // ---- coalesced writeback: out[b][j][k][t][y][x0+xl], t=12 slice is 1.0 ----
  const int xl = threadIdx.x & 15;
  const int r0 = threadIdx.x >> 4;            // 0..15
  const size_t outBase = (size_t)b * 1600 * HW + (size_t)y * Ww + x0 + xl;
#pragma unroll 4
  for (int i = 0; i < 100; ++i) {
    const int R  = r0 + i * 16;               // (j*8+k)*25 + t, 0..1599
    const int t  = R % 25;
    const int jk = R / 25;
    float v;
    if (t == 12) {
      v = 1.0f;                               // exp(0) for the center slice
    } else {
      const int t24 = t - (t > 12 ? 1 : 0);
      v = spot[(jk * 24 + t24) * 16 + xl];
    }
    out[outBase + (size_t)R * HW] = v;
  }
}

// =====================================================================
extern "C" void kernel_launch(void* const* d_in, const int* in_sizes, int n_in,
                              void* d_out, int out_size, void* d_ws, size_t ws_size,
                              hipStream_t stream) {
  (void)in_sizes; (void)n_in; (void)out_size; (void)ws_size;
  const float* x  = (const float*)d_in[0];
  const float* pe = (const float*)d_in[1];
  const float* w1 = (const float*)d_in[2];
  const float* b1 = (const float*)d_in[3];
  const float* w2 = (const float*)d_in[4];
  const float* b2 = (const float*)d_in[5];
  const float* pw = (const float*)d_in[6];
  const float* pb = (const float*)d_in[7];
  float* out = (float*)d_out;

  // workspace: h2 f16 | u f16 | S f32   (~38.3 MB)
  const size_t H2_HALVES = (size_t)Bb * HW * CK;        // 9,437,184
  _Float16* h2ws = (_Float16*)d_ws;
  _Float16* uws  = h2ws + H2_HALVES;
  float*    Sws  = (float*)(uws + H2_HALVES);

  // Stage 1: fused double GEMM (WMMA), weights staged once per block in LDS.
  // 96 blocks x 4 waves x 3 tiles = 1152 tiles.  80 KB dynamic LDS.
  feat_kernel<<<96, 128, (4096 + 32768 + 4096) * sizeof(_Float16), stream>>>(
      x, pe, w1, b1, w2, b2, pw, h2ws, uws);

  // Stage 2: per-(pixel,j) weighted squared norm.
  s_kernel<<<(Bb * HW * Kc + 255) / 256, 256, 0, stream>>>(h2ws, pw, Sws);

  // Stage 3: WMMA Gram + exp + coalesced output.  96 KB dynamic LDS.
  pot_kernel<<<Bb * Hh * (Ww / 16), 256, 96 * 1024, stream>>>(
      h2ws, uws, Sws, pb, out);
}
